// MixedScore_MultiHeadAttention_62629213110596
// MI455X (gfx1250) — matrix operand order repaired and checked by reading, not compiled, not run on hardware
//
#include <hip/hip_runtime.h>
#include <hip/hip_bf16.h>
#include <math.h>

typedef __attribute__((ext_vector_type(16))) _Float16 v16h;
typedef __attribute__((ext_vector_type(8)))  float    v8f;
typedef unsigned int u32x4 __attribute__((ext_vector_type(4)));
typedef int          i32x4 __attribute__((ext_vector_type(4)));
typedef int          i32x8 __attribute__((ext_vector_type(8)));

#define HEADS 16
#define SDIM  16
#define EMBD  256
#define MSH   16
#define BB    4
#define RR    512
#define CCC   512

// LDS strides (avoid 64-bank conflicts on row-strided access)
#define SCS   513   // f32 score row stride
#define W16S  520   // f16 weight row stride
#define ALDS_STRIDE 260  // f32 proj A-tile row stride (256 + 4 pad via TDM)

#if __has_builtin(__builtin_amdgcn_tensor_load_to_lds)
#define HAVE_TDM 1
#else
#define HAVE_TDM 0
#endif

__device__ __forceinline__ v8f wmma_f16(v16h a, v16h b, v8f c) {
  // D = A(16x32 f16) * B(32x16 f16) + C(16x16 f32)
  return __builtin_amdgcn_wmma_f32_16x16x32_f16(false, a, false, b, (short)0, c,
                                                false, false);
}

#if HAVE_TDM
// 1-D TDM copy: nelem elements of (1<<ds_code) bytes, global -> LDS.
// Optional LDS padding: pad_amount+1 DWORDs inserted every 2^(pad_interval+1)
// DWORDs stored (pad_interval=7 -> every 256 DWORDs).
__device__ __forceinline__ void tdm_copy_1d(unsigned lds_off,
                                            const void* gaddr, unsigned nelem,
                                            unsigned ds_code, int pad_en,
                                            unsigned pad_interval,
                                            unsigned pad_amount) {
  unsigned long long ga = (unsigned long long)(__SIZE_TYPE__)gaddr;
  u32x4 g0;
  g0[0] = 1u;  // count=1 valid user descriptor, no gather/restore
  g0[1] = lds_off;
  g0[2] = (unsigned)ga;
  g0[3] = ((unsigned)(ga >> 32) & 0x01FFFFFFu) | 0x80000000u;  // type=2
  i32x8 g1 = (i32x8)0;
  g1[0] = (int)((ds_code << 16) |
                (pad_en ? ((1u << 20) | (pad_interval << 22) | (pad_amount << 25))
                        : 0u));
  g1[1] = (int)((nelem & 0xFFFFu) << 16);            // tensor_dim0[15:0]
  g1[2] = (int)(((nelem >> 16) & 0xFFFFu) | (1u << 16));  // dim0 hi | tensor_dim1=1
  g1[3] = (int)((nelem & 0xFFFFu) << 16);            // tile_dim0
  g1[5] = (int)nelem;                                // tensor_dim0_stride lo
  i32x4 gz = (i32x4)0;
#if defined(__clang_major__) && (__clang_major__ >= 23)
  i32x8 gz8 = (i32x8)0;
  __builtin_amdgcn_tensor_load_to_lds(g0, g1, gz, gz, gz8, 0);
#else
  __builtin_amdgcn_tensor_load_to_lds(g0, g1, gz, gz, 0);
#endif
}
#endif

__device__ __forceinline__ void tdm_wait0() {
#if __has_builtin(__builtin_amdgcn_s_wait_tensorcnt)
  __builtin_amdgcn_s_wait_tensorcnt(0);
#elif HAVE_TDM
  asm volatile("s_wait_tensorcnt 0x0" ::: "memory");
#endif
}
__device__ __forceinline__ void tdm_wait1() {
#if __has_builtin(__builtin_amdgcn_s_wait_tensorcnt)
  __builtin_amdgcn_s_wait_tensorcnt(1);
#elif HAVE_TDM
  asm volatile("s_wait_tensorcnt 0x1" ::: "memory");
#endif
}

// ---------------------------------------------------------------------------
// Projection: out[b,h,g,s] (f16) = emb[b,g,:] @ W[:, h*16+s]
// grid = (B, G/16, 2), block = 256 (8 waves); wave w handles head bz*8+w.
// A-tile (16x256 f32) is TDM-DMA'd to LDS with bank padding; fragments gather
// from LDS. W fragments stream from global (L2-resident, coalesced).
// ---------------------------------------------------------------------------
__global__ void __launch_bounds__(256)
proj_kernel(const float* __restrict__ emb, const float* __restrict__ W,
            _Float16* __restrict__ out) {
  __shared__ __attribute__((aligned(128))) float Alds[16 * ALDS_STRIDE];

  const int tid  = threadIdx.x;
  const int lane = tid & 31;
  const int wv   = tid >> 5;
  const int hf   = lane >> 4;  // half-wave select
  const int l16  = lane & 15;
  const int b    = blockIdx.x;
  const int r0   = blockIdx.y * 16;
  const int h    = blockIdx.z * 8 + wv;
  const int n0   = h * SDIM;

  const float* Asrc = emb + ((size_t)(b * RR + r0)) * EMBD;
#if HAVE_TDM
  if (wv == 0) {
    tdm_copy_1d((unsigned)(__SIZE_TYPE__)&Alds[0], Asrc, 16 * EMBD, /*f32*/ 2,
                /*pad*/ 1, /*every 256 DW*/ 7, /*4 DW*/ 3);
    tdm_wait0();
  }
#else
  for (int i = tid; i < 16 * EMBD; i += 256)
    Alds[(i >> 8) * ALDS_STRIDE + (i & 255)] = Asrc[i];
#endif
  __syncthreads();

  v8f acc = {};
  for (int kt = 0; kt < EMBD / 32; ++kt) {
    const int k0 = kt * 32;
    v16h a, bf;
#pragma unroll
    for (int e = 0; e < 16; ++e) {
      const int k = e + hf * 16;                      // A 16x32 f16 layout
      a[e] = (_Float16)Alds[l16 * ALDS_STRIDE + k0 + k];
    }
#pragma unroll
    for (int e = 0; e < 16; ++e) {
      const int k = k0 + e + hf * 16;                 // B 32x16 f16 layout
      bf[e] = (_Float16)W[(size_t)k * (HEADS * SDIM) + n0 + l16];
    }
    acc = wmma_f16(a, bf, acc);
  }

  _Float16* obase = out + ((size_t)(b * HEADS + h)) * RR * SDIM;
#pragma unroll
  for (int j = 0; j < 8; ++j) {
    const int row = r0 + j + hf * 8;                  // C/D layout
    obase[(size_t)row * SDIM + l16] = (_Float16)acc[j];
  }
}

// ---------------------------------------------------------------------------
// Fused attention: TDM K/V -> LDS; scores(WMMA) -> per-head MLP -> exact
// softmax in LDS -> P@V (WMMA) -> out.  grid = (B*H, R/32), block = 256.
// ---------------------------------------------------------------------------
__global__ void __launch_bounds__(256)
attn_kernel(const float* __restrict__ cost_mat,
            const _Float16* __restrict__ Q, const _Float16* __restrict__ K,
            const _Float16* __restrict__ V,
            const float* __restrict__ m1w,  // [H][2][16]
            const float* __restrict__ m1b,  // [H][16]
            const float* __restrict__ m2w,  // [H][16]
            const float* __restrict__ m2b,  // [H]
            float* __restrict__ out) {      // [B][R][H*16]
  __shared__ float    sc[32 * SCS];                              // 64.1 KB
  __shared__ _Float16 w16[32 * W16S];                            // 32.5 KB
  __shared__ __attribute__((aligned(128))) _Float16 Klds[CCC * SDIM];  // 16 KB
  __shared__ __attribute__((aligned(128))) _Float16 Vlds[CCC * SDIM];  // 16 KB
  __shared__ float    pmax[32][8];
  __shared__ float    psum[32][8];
  __shared__ float    pout[4][32][16];

  const int tid  = threadIdx.x;
  const int lane = tid & 31;
  const int wv   = tid >> 5;
  const int hf   = lane >> 4;
  const int l16  = lane & 15;

  const int bh = blockIdx.x;
  const int b  = bh >> 4;
  const int h  = bh & 15;
  const int r0 = blockIdx.y * 32;

  const _Float16* Ksrc = K + (size_t)bh * CCC * SDIM;
  const _Float16* Vsrc = V + (size_t)bh * CCC * SDIM;
#if HAVE_TDM
  if (wv == 0) {  // TDM is a wave-level op; one issue each, TENSORcnt tracked
    tdm_copy_1d((unsigned)(__SIZE_TYPE__)&Klds[0], Ksrc, CCC * SDIM, /*f16*/ 1,
                0, 0, 0);
    tdm_copy_1d((unsigned)(__SIZE_TYPE__)&Vlds[0], Vsrc, CCC * SDIM, /*f16*/ 1,
                0, 0, 0);
  }
#else
  {
    const int4* ks = (const int4*)Ksrc;
    const int4* vs = (const int4*)Vsrc;
    int4* kd = (int4*)Klds;
    int4* vd = (int4*)Vlds;
    for (int i = tid; i < CCC * SDIM / 8; i += 256) {
      kd[i] = ks[i];
      vd[i] = vs[i];
    }
  }
#endif

  // Prefetch this block's cost rows (64 KB) while TDM streams K/V.
  {
    const float* cbase = cost_mat + ((size_t)(b * RR + r0)) * CCC;
    for (int i = tid; i < 512; i += 256)
      __builtin_prefetch(cbase + i * 32, 0, 3);
  }

  // Per-head MLP params into registers (uniform across block)
  float w1a[MSH], w1c[MSH], b1v[MSH], w2v[MSH];
#pragma unroll
  for (int j = 0; j < MSH; ++j) {
    w1a[j] = m1w[(h * 2 + 0) * MSH + j];
    w1c[j] = m1w[(h * 2 + 1) * MSH + j];
    b1v[j] = m1b[h * MSH + j];
    w2v[j] = m2w[h * MSH + j];
  }
  const float b2 = m2b[h];

  const int rsub = wv & 1;    // 16-row sub-block
  const int cgrp = wv >> 1;   // column-tile group (0..3)

  // Q fragment (A, 16x32 with k=16..31 zero-padded)
  const _Float16* qbase =
      Q + ((size_t)bh * RR + (r0 + rsub * 16 + l16)) * SDIM;
  v16h aq = {};
#pragma unroll
  for (int e = 0; e < 16; ++e) aq[e] = hf ? (_Float16)0 : qbase[e];  // k = e + hf*16 < 16

#if HAVE_TDM
  if (wv == 0) tdm_wait1();  // first TDM (K) complete
#endif
  __syncthreads();

  // ---- Phase 1: score tiles = MLP(QK^T/4, cost) into LDS ----
  for (int i = 0; i < 8; ++i) {
    const int c0 = (cgrp * 8 + i) * 16;
    v16h bk = {};
    if (hf == 0)  // lanes 0..15 carry k=0..15 (real); upper half zero-pad
      bk = *(const v16h*)(Klds + (c0 + l16) * SDIM);
    v8f z = {};
    v8f d = wmma_f16(aq, bk, z);
#pragma unroll
    for (int j = 0; j < 8; ++j) {
      const int rl  = rsub * 16 + j + hf * 8;
      const int col = c0 + l16;
      const float dotv  = d[j] * 0.25f;  // 1/sqrt(16)
      const float costv = cost_mat[((size_t)(b * RR + r0 + rl)) * CCC + col];
      float o = b2;
#pragma unroll
      for (int m = 0; m < MSH; ++m) {
        float hv = fmaf(w1a[m], dotv, fmaf(w1c[m], costv, b1v[m]));
        hv = fmaxf(hv, 0.0f);
        o = fmaf(w2v[m], hv, o);
      }
      sc[rl * SCS + col] = o;
    }
  }
  __syncthreads();

  // ---- Phase 2: exact softmax over 512 cols (8 threads per row) ----
  {
    const int row = tid & 31;
    const int seg = tid >> 5;
    const float* srow = sc + row * SCS + seg * 64;
    float mx = -3.4e38f;
    for (int i = 0; i < 64; ++i) mx = fmaxf(mx, srow[i]);
    pmax[row][seg] = mx;
  }
  __syncthreads();
  {
    const int row = tid & 31;
    const int seg = tid >> 5;
    float M = pmax[row][0];
#pragma unroll
    for (int s = 1; s < 8; ++s) M = fmaxf(M, pmax[row][s]);
    const float* srow = sc + row * SCS + seg * 64;
    _Float16* wrow = w16 + row * W16S + seg * 64;
    float sum = 0.0f;
    for (int i = 0; i < 64; ++i) {
      const float ev = __expf(srow[i] - M);
      sum += ev;
      wrow[i] = (_Float16)ev;
    }
    psum[row][seg] = sum;
  }
#if HAVE_TDM
  if (wv == 0) tdm_wait0();  // second TDM (V) complete before barrier
#endif
  __syncthreads();

  // ---- Phase 3: partial P@V, K=512 split across 4 wave-quads ----
  {
    const int kq = wv >> 1;
    v8f acc = {};
    for (int kt = 0; kt < 4; ++kt) {
      const int c0 = kq * 128 + kt * 32;
      v16h a, bv;
#pragma unroll
      for (int e = 0; e < 16; ++e) {
        const int k = e + hf * 16;
        a[e] = w16[(rsub * 16 + l16) * W16S + c0 + k];
      }
#pragma unroll
      for (int e = 0; e < 16; ++e) {
        const int k = c0 + e + hf * 16;
        bv[e] = Vlds[k * SDIM + l16];
      }
      acc = wmma_f16(a, bv, acc);
    }
#pragma unroll
    for (int j = 0; j < 8; ++j)
      pout[kq][rsub * 16 + j + hf * 8][l16] = acc[j];
  }
  __syncthreads();

  // ---- Phase 4: reduce quads, apply 1/L, store [b, r, h*16+s] ----
  for (int e = tid; e < 32 * 16; e += 256) {
    const int row = e >> 4;
    const int s   = e & 15;
    const float v = pout[0][row][s] + pout[1][row][s] +
                    pout[2][row][s] + pout[3][row][s];
    float L = 0.0f;
#pragma unroll
    for (int q = 0; q < 8; ++q) L += psum[row][q];
    out[((size_t)(b * RR + r0 + row)) * (HEADS * SDIM) + h * SDIM + s] = v / L;
  }
}

// ---------------------------------------------------------------------------
extern "C" void kernel_launch(void* const* d_in, const int* in_sizes, int n_in,
                              void* d_out, int out_size, void* d_ws,
                              size_t ws_size, hipStream_t stream) {
  const float* row_emb = (const float*)d_in[0];
  const float* col_emb = (const float*)d_in[1];
  const float* cost    = (const float*)d_in[2];
  const float* Wq      = (const float*)d_in[3];
  const float* Wk      = (const float*)d_in[4];
  const float* Wv      = (const float*)d_in[5];
  const float* m1w     = (const float*)d_in[6];
  const float* m1b     = (const float*)d_in[7];
  const float* m2w     = (const float*)d_in[8];
  const float* m2b     = (const float*)d_in[9];
  float* out = (float*)d_out;

  const size_t per = (size_t)BB * HEADS * RR * SDIM;  // 512K halves = 1 MB
  _Float16* Qws = (_Float16*)d_ws;
  _Float16* Kws = Qws + per;
  _Float16* Vws = Kws + per;

  dim3 pgrid(BB, RR / 16, 2);
  proj_kernel<<<pgrid, 256, 0, stream>>>(row_emb, Wq, Qws);
  proj_kernel<<<pgrid, 256, 0, stream>>>(col_emb, Wk, Kws);
  proj_kernel<<<pgrid, 256, 0, stream>>>(col_emb, Wv, Vws);

  attn_kernel<<<dim3(BB * HEADS, RR / 32), 256, 0, stream>>>(
      cost, Qws, Kws, Vws, m1w, m1b, m2w, m2b, out);
}
